// CoevolExtractor_14216341750247
// MI455X (gfx1250) — matrix operand order repaired and checked by reading, not compiled, not run on hardware
//
#include <hip/hip_runtime.h>
#include <hip/hip_bf16.h>

// ---- CDNA5 WMMA types --------------------------------------------------
typedef __attribute__((ext_vector_type(16))) __bf16 v16bf;
typedef __attribute__((ext_vector_type(8)))  float  v8f;

union BfOp { v16bf v; unsigned int u[8]; };

__device__ __forceinline__ unsigned short f2bf(float f) {
  // round-to-nearest-even fp32 -> bf16
  unsigned int u = __float_as_uint(f);
  unsigned int r = u + 0x7FFFu + ((u >> 16) & 1u);
  return (unsigned short)(r >> 16);
}

__device__ __forceinline__ unsigned int f2bf_pack(float lo, float hi) {
  return (unsigned int)f2bf(lo) | ((unsigned int)f2bf(hi) << 16);
}

__device__ __forceinline__ v8f wmma_bf16(v16bf a, v16bf b, v8f c) {
  return __builtin_amdgcn_wmma_f32_16x16x32_bf16(
      /*neg_a=*/false, a, /*neg_b=*/false, b,
      /*c_mod=*/(short)0, c, /*reuse_a=*/false, /*reuse_b=*/false);
}

// ---- problem constants (B=1, N=128, L=256, P=32, n_out=128) ------------
#define NN      128           // contraction depth (MSA axis)
#define LLEN    256           // sequence length
#define PDIM    32            // per-position channels
#define NOUT    128           // output features
#define CH      1024          // P*P layernorm channels

#define AB_ROW  132           // padded LDS row stride (ushorts) for 128-deep rows
#define AB_BLK  (32 * AB_ROW) // one (i or l) slot: 32 rows x 132
#define PLN_ROW 1032          // padded row stride for normalized 16x1024 block

// A-operand (16-bit, 16x32): lane m16 = M row; K runs {8g..8g+7} and {16+8g..23+8g}
__device__ __forceinline__ void load_opA(const unsigned int* lds, int baseUshort, BfOp& op) {
  int b = baseUshort >> 1;
#pragma unroll
  for (int t = 0; t < 4; ++t) op.u[t] = lds[b + t];
#pragma unroll
  for (int t = 0; t < 4; ++t) op.u[4 + t] = lds[b + 8 + t];
}

// B-operand (16-bit, 32x16): lane m16 = N col; K runs {8g..8g+7} and {16+8g..23+8g}
__device__ __forceinline__ void load_opB(const unsigned int* lds, int baseUshort, BfOp& op) {
  int b = baseUshort >> 1;
#pragma unroll
  for (int t = 0; t < 4; ++t) op.u[t] = lds[b + t];
#pragma unroll
  for (int t = 0; t < 4; ++t) op.u[4 + t] = lds[b + 8 + t];
}

__device__ __forceinline__ v8f vzero8() {
  v8f z;
#pragma unroll
  for (int r = 0; r < 8; ++r) z[r] = 0.0f;
  return z;
}

__launch_bounds__(256)
__global__ void coevol_fused_kernel(const float* __restrict__ xd,
                                    const float* __restrict__ xw,
                                    const float* __restrict__ gamma,
                                    const float* __restrict__ beta,
                                    const float* __restrict__ Wg,
                                    const float* __restrict__ bg,
                                    float* __restrict__ outp)
{
  // regionA: phase1 A-tiles (bf16, transposed [di][j][n]); later holds LN'ed pln[16][1024]
  // regionB: phase1 B-tiles (bf16, transposed [dl][m][n]); later holds W slice [128][128]
  __shared__ unsigned short smA[4 * AB_BLK];   // 33792 B
  __shared__ unsigned short smB[4 * AB_BLK];   // 33792 B
  unsigned int*       smA32  = (unsigned int*)smA;
  unsigned int*       smB32  = (unsigned int*)smB;
  const unsigned int* lds32A = (const unsigned int*)smA;
  const unsigned int* lds32B = (const unsigned int*)smB;

  const int t    = threadIdx.x;
  const int w    = t >> 5;          // wave id 0..7
  const int lane = t & 31;
  const int g    = lane >> 4;       // lane-half
  const int m16  = lane & 15;

  const int i0 = blockIdx.y * 4;
  const int l0 = blockIdx.x * 4;

  // ---------- stage x_down / x_down_w tiles into LDS, transposed, bf16 ----
  // pack two consecutive-n bf16 per 32-bit DS store
  for (int idx = t; idx < 4 * PDIM * (NN / 2); idx += 256) {
    int di  = idx >> 11;            // 0..3
    int rem = idx & 2047;
    int np  = rem >> 5;             // n-pair 0..63
    int j   = rem & 31;
    size_t gbase = (size_t)(2 * np) * (LLEN * PDIM) + (size_t)(i0 + di) * PDIM + j;
    unsigned int ua = f2bf_pack(xd[gbase], xd[gbase + (size_t)(LLEN * PDIM)]);
    smA32[((di * AB_BLK + j * AB_ROW) >> 1) + np] = ua;
    size_t gbaseB = (size_t)(2 * np) * (LLEN * PDIM) + (size_t)(l0 + di) * PDIM + j;
    unsigned int ub = f2bf_pack(xw[gbaseB], xw[gbaseB + (size_t)(LLEN * PDIM)]);
    smB32[((di * AB_BLK + j * AB_ROW) >> 1) + np] = ub;
  }
  __syncthreads();

  // ---------- phase 1: pair(i,l) = A_i^T (32x128) @ B_l (128x32) ----------
  // wave owns two positions sharing di: p = di*4 + dl0 + c, c in {0,1}
  //   di = w>>1, dl0 = (w&1)*2  -> A operands shared across both positions
  const int di  = w >> 1;
  const int dl0 = (w & 1) * 2;

  v8f acc[2][2][2];                 // [c][jt][mt]
#pragma unroll
  for (int c = 0; c < 2; ++c)
#pragma unroll
    for (int jt = 0; jt < 2; ++jt)
#pragma unroll
      for (int mt = 0; mt < 2; ++mt) acc[c][jt][mt] = vzero8();

#pragma unroll
  for (int nc = 0; nc < 4; ++nc) {          // K = 128 in chunks of 32
    BfOp a0, a1;
    load_opA(lds32A, di * AB_BLK + (m16)      * AB_ROW + nc * 32 + 8 * g, a0);
    load_opA(lds32A, di * AB_BLK + (16 + m16) * AB_ROW + nc * 32 + 8 * g, a1);
#pragma unroll
    for (int c = 0; c < 2; ++c) {
      const int dl = dl0 + c;
      BfOp b0, b1;
      load_opB(lds32B, dl * AB_BLK + (m16)      * AB_ROW + nc * 32 + 8 * g, b0);
      load_opB(lds32B, dl * AB_BLK + (16 + m16) * AB_ROW + nc * 32 + 8 * g, b1);
      acc[c][0][0] = wmma_bf16(a0.v, b0.v, acc[c][0][0]);
      acc[c][0][1] = wmma_bf16(a0.v, b1.v, acc[c][0][1]);
      acc[c][1][0] = wmma_bf16(a1.v, b0.v, acc[c][1][0]);
      acc[c][1][1] = wmma_bf16(a1.v, b1.v, acc[c][1][1]);
    }
  }
  __syncthreads();   // phase-1 LDS reads complete before regions are reused

  // ---------- LayerNorm over the 1024 channels of each position ----------
#pragma unroll
  for (int c = 0; c < 2; ++c) {
    float s1 = 0.0f, s2 = 0.0f;
#pragma unroll
    for (int jt = 0; jt < 2; ++jt)
#pragma unroll
      for (int mt = 0; mt < 2; ++mt)
#pragma unroll
        for (int r = 0; r < 8; ++r) {
          float v = acc[c][jt][mt][r];
          s1 += v; s2 += v * v;
        }
#pragma unroll
    for (int off = 16; off > 0; off >>= 1) {
      s1 += __shfl_xor(s1, off, 32);
      s2 += __shfl_xor(s2, off, 32);
    }
    const float mu   = s1 * (1.0f / CH);
    const float var  = s2 * (1.0f / CH) - mu * mu;
    const float rstd = rsqrtf(var + 1e-5f);

    const int p = di * 4 + dl0 + c;         // global position index 0..15
#pragma unroll
    for (int jt = 0; jt < 2; ++jt)
#pragma unroll
      for (int mt = 0; mt < 2; ++mt)
#pragma unroll
        for (int r = 0; r < 8; ++r) {
          int j  = jt * 16 + 8 * g + r;     // C/D layout: M = r + 8g
          int m  = mt * 16 + m16;           // N = lane%16
          int ch = j * 32 + m;
          float v = (acc[c][jt][mt][r] - mu) * rstd * gamma[ch] + beta[ch];
          smA[p * PLN_ROW + ch] = f2bf(v);  // pln[16][1024] (padded), reuses regionA
        }
  }
  __syncthreads();

  // ---------- phase 2: out(16 pos x 128) = pln (16x1024) @ W^T ------------
  // wave w owns output o-tile [16w, 16w+16); K=1024 streamed as 8 LDS slices
  v8f oacc = vzero8();
  for (int s = 0; s < 8; ++s) {
    // stage W[:, s*128 .. s*128+128) -> bf16 LDS slice, packed b32 stores
    for (int idx = t; idx < NOUT * 64; idx += 256) {
      int o  = idx >> 6;
      int kp = idx & 63;                    // k-pair
      const float2 wv = *(const float2*)(Wg + (size_t)o * CH + s * 128 + 2 * kp);
      smB32[o * (AB_ROW / 2) + kp] = f2bf_pack(wv.x, wv.y);
    }
    __syncthreads();

    if (s < 7) {
      // prefetch next slice while this slice's WMMAs run (global_prefetch_b8)
      const float* pf = Wg + (size_t)(t >> 1) * CH + (s + 1) * 128 + (t & 1) * 64;
      __builtin_prefetch(pf, 0, 1);
    }

#pragma unroll
    for (int kc = 0; kc < 4; ++kc) {
      BfOp av, bw;
      load_opA(lds32A, m16 * PLN_ROW + s * 128 + kc * 32 + 8 * g,  av);
      load_opB(lds32B, (w * 16 + m16) * AB_ROW + kc * 32 + 8 * g, bw);
      oacc = wmma_bf16(av.v, bw.v, oacc);
    }
    __syncthreads();
  }

  // ---------- bias + store ------------------------------------------------
  const int o = w * 16 + m16;
  const float bias = bg[o];
#pragma unroll
  for (int r = 0; r < 8; ++r) {
    int p = r + 8 * g;                 // D layout: M = r + 8g -> position
    int i = i0 + (p >> 2);
    int l = l0 + (p & 3);
    outp[((size_t)(i * LLEN + l)) * NOUT + o] = oacc[r] + bias;
  }
}

extern "C" void kernel_launch(void* const* d_in, const int* in_sizes, int n_in,
                              void* d_out, int out_size, void* d_ws, size_t ws_size,
                              hipStream_t stream) {
  (void)in_sizes; (void)n_in; (void)out_size; (void)d_ws; (void)ws_size;
  const float* xd    = (const float*)d_in[0];  // x_down  (1,128,256,32)
  const float* xw    = (const float*)d_in[1];  // x_down_w(1,128,256,32)
  const float* gamma = (const float*)d_in[2];  // (1024,)
  const float* beta  = (const float*)d_in[3];  // (1024,)
  const float* Wg    = (const float*)d_in[4];  // (128,1024)
  const float* bg    = (const float*)d_in[5];  // (128,)
  float* outp = (float*)d_out;                 // (1,256,256,128) fp32

  dim3 grid(LLEN / 4, LLEN / 4, 1);   // 64 x 64 workgroups, each a 4x4 (i,l) block
  dim3 block(256, 1, 1);              // 8 waves (wave32)
  hipLaunchKernelGGL(coevol_fused_kernel, grid, block, 0, stream,
                     xd, xw, gamma, beta, Wg, bg, outp);
}
